// WanAttentionBlock_43937515438492
// MI455X (gfx1250) — matrix operand order repaired and checked by reading, not compiled, not run on hardware
//
#include <hip/hip_runtime.h>
#include <hip/hip_bf16.h>
#include <math.h>

typedef __bf16 bf16;
typedef __attribute__((ext_vector_type(16))) __bf16 bf16x16;
typedef __attribute__((ext_vector_type(8)))  float  f32x8;

union FragA { bf16x16 v; bf16 e[16]; int4 q[2]; };
union FragC { f32x8  v; float f[8]; };

#define C_DIM   1536
#define L_DIM   2400
#define L2_DIM  512
#define FF_DIM  8960
#define N_HEADS 12
#define D_HEAD  128
#define F_G 2
#define H_G 30
#define W_G 40
#define EPSF 1e-6f
#define ATT_SCALE 0.08838834764831845f  // 1/sqrt(128)

// ---------------------------------------------------------------------------
// fp32 -> bf16 conversion (grid-stride)
// ---------------------------------------------------------------------------
__global__ __launch_bounds__(256) void cvt_f32_bf16(const float* __restrict__ in,
                                                    bf16* __restrict__ out, int n) {
  for (int i = blockIdx.x * blockDim.x + threadIdx.x; i < n; i += gridDim.x * blockDim.x)
    out[i] = (bf16)in[i];
}

// ---------------------------------------------------------------------------
// fp32 [R x N] -> bf16 transposed [N x R], 32x32 LDS tiles
// ---------------------------------------------------------------------------
__global__ __launch_bounds__(256) void cvt_transpose_f32_bf16(const float* __restrict__ in,
                                                              bf16* __restrict__ out,
                                                              int R, int N) {
  __shared__ bf16 tile[32][34];
  const int tx = threadIdx.x & 31;
  const int ty = threadIdx.x >> 5;        // 0..7
  const int r0 = blockIdx.y * 32;
  const int c0 = blockIdx.x * 32;
#pragma unroll
  for (int j = 0; j < 4; ++j)
    tile[ty + j * 8][tx] = (bf16)in[(size_t)(r0 + ty + j * 8) * N + c0 + tx];
  __syncthreads();
#pragma unroll
  for (int j = 0; j < 4; ++j)
    out[(size_t)(c0 + ty + j * 8) * R + r0 + tx] = tile[tx][ty + j * 8];
}

// ---------------------------------------------------------------------------
// h = shift + LN(x) * (1 + scale), shift = mod[is]+e[is], scale = mod[ic]+e[ic]
// ---------------------------------------------------------------------------
__global__ __launch_bounds__(256) void ln_mod_kernel(const float* __restrict__ x,
                                                     const float* __restrict__ e6,
                                                     const float* __restrict__ mod6,
                                                     int is, int ic,
                                                     bf16* __restrict__ out) {
  const int row = blockIdx.x;
  const int tid = threadIdx.x;
  __shared__ float rs[256], rq[256];
  float s1 = 0.f, s2 = 0.f;
  for (int c = tid; c < C_DIM; c += 256) {
    float v = x[(size_t)row * C_DIM + c];
    s1 += v; s2 += v * v;
  }
  rs[tid] = s1; rq[tid] = s2; __syncthreads();
  for (int st = 128; st > 0; st >>= 1) {
    if (tid < st) { rs[tid] += rs[tid + st]; rq[tid] += rq[tid + st]; }
    __syncthreads();
  }
  const float mean = rs[0] * (1.f / C_DIM);
  const float var  = rq[0] * (1.f / C_DIM) - mean * mean;
  const float rstd = rsqrtf(var + EPSF);
  for (int c = tid; c < C_DIM; c += 256) {
    float shift = mod6[is * C_DIM + c] + e6[is * C_DIM + c];
    float scale = 1.f + mod6[ic * C_DIM + c] + e6[ic * C_DIM + c];
    float v = (x[(size_t)row * C_DIM + c] - mean) * rstd;
    out[(size_t)row * C_DIM + c] = (bf16)(shift + v * scale);
  }
}

// ---------------------------------------------------------------------------
// RMS-norm over C (+ optional Wan RoPE), fp32 in -> bf16 out. One block/row.
// mode bit0: rms, bit1: rope
// ---------------------------------------------------------------------------
__global__ __launch_bounds__(256) void rms_rope_kernel(const float* __restrict__ in,
                                                       const float* __restrict__ w,
                                                       const float* __restrict__ freqs,
                                                       bf16* __restrict__ out,
                                                       int mode) {
  const int row = blockIdx.x;
  const int tid = threadIdx.x;
  __shared__ float rq[256];
  float s2 = 0.f;
  for (int c = tid; c < C_DIM; c += 256) {
    float v = in[(size_t)row * C_DIM + c];
    s2 += v * v;
  }
  rq[tid] = s2; __syncthreads();
  for (int st = 128; st > 0; st >>= 1) {
    if (tid < st) rq[tid] += rq[tid + st];
    __syncthreads();
  }
  const float rn = rsqrtf(rq[0] * (1.f / C_DIM) + EPSF);

  if (mode & 2) {
    const int f  = row / (H_G * W_G);
    const int rm = row % (H_G * W_G);
    const int hh = rm / W_G;
    const int ww = rm % W_G;
    for (int j = tid; j < (C_DIM / 2); j += 256) {
      const int head = j >> 6;        // D/2 = 64 pairs per head
      const int jj   = j & 63;
      const int c0   = head * D_HEAD + 2 * jj;
      int idx = (jj < 22) ? f : ((jj < 43) ? hh : ww);
      const float cs = freqs[((size_t)idx * 64 + jj) * 2 + 0];
      const float sn = freqs[((size_t)idx * 64 + jj) * 2 + 1];
      float xr = in[(size_t)row * C_DIM + c0]     * rn * w[c0];
      float xi = in[(size_t)row * C_DIM + c0 + 1] * rn * w[c0 + 1];
      out[(size_t)row * C_DIM + c0]     = (bf16)(xr * cs - xi * sn);
      out[(size_t)row * C_DIM + c0 + 1] = (bf16)(xr * sn + xi * cs);
    }
  } else {
    for (int c = tid; c < C_DIM; c += 256) {
      float v = in[(size_t)row * C_DIM + c];
      if (mode & 1) v = v * rn * w[c];
      out[(size_t)row * C_DIM + c] = (bf16)v;
    }
  }
}

// ---------------------------------------------------------------------------
// residual: out = xin + y * s, s = mod[idx]+e[idx] (idx>=0) else 1
// ---------------------------------------------------------------------------
__global__ __launch_bounds__(256) void resid_kernel(const float* __restrict__ xin,
                                                    const float* __restrict__ y,
                                                    const float* __restrict__ e6,
                                                    const float* __restrict__ mod6,
                                                    int idx, float* __restrict__ xout,
                                                    int n) {
  for (int i = blockIdx.x * blockDim.x + threadIdx.x; i < n; i += gridDim.x * blockDim.x) {
    int c = i % C_DIM;
    float s = (idx >= 0) ? (mod6[idx * C_DIM + c] + e6[idx * C_DIM + c]) : 1.f;
    xout[i] = xin[i] + y[i] * s;
  }
}

// ---------------------------------------------------------------------------
// Tiled bf16 WMMA GEMM: C[MxN] = A[MxK] * Bt^T (+bias), Bt is [N x K].
// 128x128 block tile, 8 waves each 64x32 (4x2 wmma accs), K staged 64/iter.
// All fragment LDS reads are contiguous 2x b128 per lane.
// epi: 0 -> f32 out, 1 -> bf16 out, 2 -> bf16 gelu(out)
// ---------------------------------------------------------------------------
#define GBM 128
#define GBN 128
#define GBK 64
#define KSTRIDE (GBK + 8)   // 72 elems -> 144B rows (16B aligned)

__device__ __forceinline__ float gelu_tanh(float v) {
  return 0.5f * v * (1.f + tanhf(0.7978845608028654f * (v + 0.044715f * v * v * v)));
}

__global__ __launch_bounds__(256) void gemm_bf16_kernel(const bf16* __restrict__ A,
                                                        const bf16* __restrict__ Bt,
                                                        const float* __restrict__ bias,
                                                        float* __restrict__ outF,
                                                        bf16* __restrict__ outB,
                                                        int M, int N, int K, int epi) {
  __shared__ bf16 As[GBM][KSTRIDE];
  __shared__ bf16 Bs[GBN][KSTRIDE];

  const int tid  = threadIdx.x;
  const int wave = tid >> 5;
  const int lane = tid & 31;
  const int lh   = lane >> 4;
  const int ln   = lane & 15;

  const int tileN = blockIdx.x * GBN;
  const int tileM = blockIdx.y * GBM;

  const int waveM = wave >> 2;   // 0..1  (64 rows each)
  const int waveN = wave & 3;    // 0..3  (32 cols each)

  FragC acc[4][2];
#pragma unroll
  for (int mi = 0; mi < 4; ++mi)
#pragma unroll
    for (int ni = 0; ni < 2; ++ni)
#pragma unroll
      for (int r = 0; r < 8; ++r) acc[mi][ni].f[r] = 0.f;

  const int srow = tid >> 1, sseg = tid & 1;   // 128 rows, 2 x 32-elem segments
  const int4 z4 = make_int4(0, 0, 0, 0);

  for (int k0 = 0; k0 < K; k0 += GBK) {
    // stage A tile (M x 64)
    {
      const int gr = tileM + srow;
      int4* dst = (int4*)&As[srow][sseg * 32];
      if (gr < M) {
        const bf16* sp = A + (size_t)gr * K + k0 + sseg * 32;
        if (k0 + GBK < K) __builtin_prefetch(sp + GBK, 0, 0);
        const int4* src = (const int4*)sp;
#pragma unroll
        for (int i = 0; i < 4; ++i) dst[i] = src[i];
      } else {
#pragma unroll
        for (int i = 0; i < 4; ++i) dst[i] = z4;
      }
    }
    // stage B tile (128 n-rows x 64 k) from transposed weights
    {
      const bf16* sp = Bt + (size_t)(tileN + srow) * K + k0 + sseg * 32;
      if (k0 + GBK < K) __builtin_prefetch(sp + GBK, 0, 0);
      const int4* src = (const int4*)sp;
      int4* dst = (int4*)&Bs[srow][sseg * 32];
#pragma unroll
      for (int i = 0; i < 4; ++i) dst[i] = src[i];
    }
    __syncthreads();

#pragma unroll
    for (int ks = 0; ks < 2; ++ks) {
      const int kb = ks * 32;
      FragA a[4], b[2];
#pragma unroll
      for (int mi = 0; mi < 4; ++mi) {
        const int m = waveM * 64 + mi * 16 + ln;
        a[mi].q[0] = *(const int4*)&As[m][kb + lh * 8];
        a[mi].q[1] = *(const int4*)&As[m][kb + 16 + lh * 8];
      }
#pragma unroll
      for (int ni = 0; ni < 2; ++ni) {
        const int n = waveN * 32 + ni * 16 + ln;
        b[ni].q[0] = *(const int4*)&Bs[n][kb + lh * 8];
        b[ni].q[1] = *(const int4*)&Bs[n][kb + 16 + lh * 8];
      }
#pragma unroll
      for (int mi = 0; mi < 4; ++mi)
#pragma unroll
        for (int ni = 0; ni < 2; ++ni)
          acc[mi][ni].v = __builtin_amdgcn_wmma_f32_16x16x32_bf16(
              false, a[mi].v, false, b[ni].v, (short)0, acc[mi][ni].v, false, false);
    }
    __syncthreads();
  }

  // epilogue
#pragma unroll
  for (int mi = 0; mi < 4; ++mi) {
#pragma unroll
    for (int ni = 0; ni < 2; ++ni) {
      const int col = tileN + waveN * 32 + ni * 16 + ln;
      const float bv = bias ? bias[col] : 0.f;
#pragma unroll
      for (int r = 0; r < 8; ++r) {
        const int row = tileM + waveM * 64 + mi * 16 + r + 8 * lh;
        if (row < M) {
          float v = acc[mi][ni].f[r] + bv;
          if (epi == 0)      outF[(size_t)row * N + col] = v;
          else if (epi == 1) outB[(size_t)row * N + col] = (bf16)v;
          else               outB[(size_t)row * N + col] = (bf16)gelu_tanh(v);
        }
      }
    }
  }
}

// ---------------------------------------------------------------------------
// Flash attention: 4 waves/block, each wave owns 16 q rows, full D=128.
// K staged row-major (kv x d); V staged TRANSPOSED (d x kv) so that both
// QK^T and PV B-fragments are contiguous b128 LDS reads.
// ---------------------------------------------------------------------------
#define KVSTR (D_HEAD + 8)   // 136 elems, 272B rows
#define VTSTR (64 + 8)       // 72 elems, 144B rows
#define PSTR  (64 + 8)

__global__ __launch_bounds__(128) void flash_kernel(const bf16* __restrict__ Q,
                                                    const bf16* __restrict__ Km,
                                                    const bf16* __restrict__ Vm,
                                                    float* __restrict__ O,
                                                    int Sq, int Sk) {
  __shared__ bf16 Ks[64][KVSTR];          // [kv][d]
  __shared__ bf16 Vt[D_HEAD][VTSTR];      // [d][kv]  (transposed)
  __shared__ bf16 Ps[4][16][PSTR];

  const int tid  = threadIdx.x;
  const int wave = tid >> 5;
  const int lane = tid & 31;
  const int lh   = lane >> 4;
  const int ln   = lane & 15;
  const int head = blockIdx.y;
  const int qBase = blockIdx.x * 64 + wave * 16;

  // preload Q fragments (A layout: m = ln, K split over lane-half)
  FragA aq[4];
  {
    int qr = qBase + ln; if (qr >= Sq) qr = Sq - 1;
    const bf16* qp = Q + (size_t)qr * C_DIM + head * D_HEAD;
#pragma unroll
    for (int dt = 0; dt < 4; ++dt) {
      aq[dt].q[0] = *(const int4*)(qp + dt * 32 + lh * 8);
      aq[dt].q[1] = *(const int4*)(qp + dt * 32 + 16 + lh * 8);
    }
  }

  FragC o[8];
  float mprev[8], lacc[8];
#pragma unroll
  for (int dt = 0; dt < 8; ++dt)
#pragma unroll
    for (int r = 0; r < 8; ++r) o[dt].f[r] = 0.f;
#pragma unroll
  for (int r = 0; r < 8; ++r) { mprev[r] = -1e30f; lacc[r] = 0.f; }

  const int lrow = tid >> 1, lseg = tid & 1;   // KV tile load split
  const int4 z4 = make_int4(0, 0, 0, 0);

  for (int kv0 = 0; kv0 < Sk; kv0 += 64) {
    __syncthreads();
    {
      const int kvr = kv0 + lrow;
      int4* dk = (int4*)&Ks[lrow][lseg * 64];
      if (kvr < Sk) {
        const int4* sk = (const int4*)(Km + (size_t)kvr * C_DIM + head * D_HEAD + lseg * 64);
        const int4* sv = (const int4*)(Vm + (size_t)kvr * C_DIM + head * D_HEAD + lseg * 64);
#pragma unroll
        for (int i = 0; i < 8; ++i) {
          dk[i] = sk[i];
          int4 vv = sv[i];
          const bf16* vb = (const bf16*)&vv;
#pragma unroll
          for (int j = 0; j < 8; ++j)
            Vt[lseg * 64 + i * 8 + j][lrow] = vb[j];   // transpose into LDS
        }
      } else {
#pragma unroll
        for (int i = 0; i < 8; ++i) {
          dk[i] = z4;
#pragma unroll
          for (int j = 0; j < 8; ++j)
            Vt[lseg * 64 + i * 8 + j][lrow] = (bf16)0.f;
        }
      }
    }
    __syncthreads();

    // S = Q * K^T  (16 x 64)
    FragC s[4];
#pragma unroll
    for (int nt = 0; nt < 4; ++nt)
#pragma unroll
      for (int r = 0; r < 8; ++r) s[nt].f[r] = 0.f;

#pragma unroll
    for (int dt = 0; dt < 4; ++dt) {
#pragma unroll
      for (int nt = 0; nt < 4; ++nt) {
        FragA b;
        b.q[0] = *(const int4*)&Ks[nt * 16 + ln][dt * 32 + lh * 8];
        b.q[1] = *(const int4*)&Ks[nt * 16 + ln][dt * 32 + 16 + lh * 8];
        s[nt].v = __builtin_amdgcn_wmma_f32_16x16x32_bf16(
            false, aq[dt].v, false, b.v, (short)0, s[nt].v, false, false);
      }
    }

    // scale + mask + online softmax
    float mloc[8];
#pragma unroll
    for (int r = 0; r < 8; ++r) mloc[r] = -1e30f;
#pragma unroll
    for (int nt = 0; nt < 4; ++nt) {
      const bool valid = (kv0 + nt * 16 + ln) < Sk;
#pragma unroll
      for (int r = 0; r < 8; ++r) {
        float v = s[nt].f[r] * ATT_SCALE;
        v = valid ? v : -1e30f;
        s[nt].f[r] = v;
        mloc[r] = fmaxf(mloc[r], v);
      }
    }
#pragma unroll
    for (int r = 0; r < 8; ++r) {
#pragma unroll
      for (int off = 1; off < 16; off <<= 1)
        mloc[r] = fmaxf(mloc[r], __shfl_xor(mloc[r], off, 16));
      const float mn = fmaxf(mprev[r], mloc[r]);
      const float alpha = expf(mprev[r] - mn);
      mprev[r] = mn;
      lacc[r] *= alpha;
#pragma unroll
      for (int dt = 0; dt < 8; ++dt) o[dt].f[r] *= alpha;
    }

    float lloc[8];
#pragma unroll
    for (int r = 0; r < 8; ++r) lloc[r] = 0.f;
#pragma unroll
    for (int nt = 0; nt < 4; ++nt)
#pragma unroll
      for (int r = 0; r < 8; ++r) {
        const float p = expf(s[nt].f[r] - mprev[r]);
        lloc[r] += p;
        Ps[wave][r + 8 * lh][nt * 16 + ln] = (bf16)p;
      }
#pragma unroll
    for (int r = 0; r < 8; ++r) {
#pragma unroll
      for (int off = 1; off < 16; off <<= 1)
        lloc[r] += __shfl_xor(lloc[r], off, 16);
      lacc[r] += lloc[r];
    }

    // O += P * V   (16 x 128), k-dim = 64 in 2 steps; V from transposed LDS
#pragma unroll
    for (int ks = 0; ks < 2; ++ks) {
      FragA a;
      a.q[0] = *(const int4*)&Ps[wave][ln][ks * 32 + lh * 8];
      a.q[1] = *(const int4*)&Ps[wave][ln][ks * 32 + 16 + lh * 8];
#pragma unroll
      for (int dt = 0; dt < 8; ++dt) {
        FragA b;
        b.q[0] = *(const int4*)&Vt[dt * 16 + ln][ks * 32 + lh * 8];
        b.q[1] = *(const int4*)&Vt[dt * 16 + ln][ks * 32 + 16 + lh * 8];
        o[dt].v = __builtin_amdgcn_wmma_f32_16x16x32_bf16(
            false, a.v, false, b.v, (short)0, o[dt].v, false, false);
      }
    }
  }

  // write O / l
#pragma unroll
  for (int r = 0; r < 8; ++r) {
    const int row = qBase + r + 8 * lh;
    if (row < Sq) {
      const float invl = (lacc[r] > 0.f) ? (1.f / lacc[r]) : 0.f;
#pragma unroll
      for (int dt = 0; dt < 8; ++dt)
        O[(size_t)row * C_DIM + head * D_HEAD + dt * 16 + ln] = o[dt].f[r] * invl;
    }
  }
}

// ---------------------------------------------------------------------------
// host orchestration
// ---------------------------------------------------------------------------
static inline int cdiv(int a, int b) { return (a + b - 1) / b; }

extern "C" void kernel_launch(void* const* d_in, const int* in_sizes, int n_in,
                              void* d_out, int out_size, void* d_ws, size_t ws_size,
                              hipStream_t stream) {
  (void)in_sizes; (void)n_in; (void)out_size; (void)ws_size;
  const float* x      = (const float*)d_in[0];
  const float* e6     = (const float*)d_in[1];
  const float* ctx    = (const float*)d_in[2];
  const float* freqs  = (const float*)d_in[3];
  const float* mod6   = (const float*)d_in[7];
  const float* sa_q_w = (const float*)d_in[8];  const float* sa_q_b = (const float*)d_in[9];
  const float* sa_k_w = (const float*)d_in[10]; const float* sa_k_b = (const float*)d_in[11];
  const float* sa_v_w = (const float*)d_in[12]; const float* sa_v_b = (const float*)d_in[13];
  const float* sa_o_w = (const float*)d_in[14]; const float* sa_o_b = (const float*)d_in[15];
  const float* sa_nq  = (const float*)d_in[16]; const float* sa_nk  = (const float*)d_in[17];
  const float* ca_q_w = (const float*)d_in[18]; const float* ca_q_b = (const float*)d_in[19];
  const float* ca_k_w = (const float*)d_in[20]; const float* ca_k_b = (const float*)d_in[21];
  const float* ca_v_w = (const float*)d_in[22]; const float* ca_v_b = (const float*)d_in[23];
  const float* ca_o_w = (const float*)d_in[24]; const float* ca_o_b = (const float*)d_in[25];
  const float* ca_nq  = (const float*)d_in[26]; const float* ca_nk  = (const float*)d_in[27];
  const float* ffn_w1 = (const float*)d_in[28]; const float* ffn_b1 = (const float*)d_in[29];
  const float* ffn_w2 = (const float*)d_in[30]; const float* ffn_b2 = (const float*)d_in[31];
  float* outp = (float*)d_out;

  char* base = (char*)d_ws;
  size_t off = 0;
  auto alloc = [&](size_t bytes) -> void* {
    void* p = base + off;
    off = (off + bytes + 255) & ~(size_t)255;
    return p;
  };
  const size_t CC = (size_t)C_DIM * C_DIM;
  bf16* w_saq = (bf16*)alloc(CC * 2);  bf16* w_sak = (bf16*)alloc(CC * 2);
  bf16* w_sav = (bf16*)alloc(CC * 2);  bf16* w_sao = (bf16*)alloc(CC * 2);
  bf16* w_caq = (bf16*)alloc(CC * 2);  bf16* w_cak = (bf16*)alloc(CC * 2);
  bf16* w_cav = (bf16*)alloc(CC * 2);  bf16* w_cao = (bf16*)alloc(CC * 2);
  bf16* w_ff1 = (bf16*)alloc((size_t)C_DIM * FF_DIM * 2);   // stored [FF x C]
  bf16* w_ff2 = (bf16*)alloc((size_t)FF_DIM * C_DIM * 2);   // stored [C x FF]
  bf16* Hb = (bf16*)alloc((size_t)L_DIM * C_DIM * 2);
  bf16* Qb = (bf16*)alloc((size_t)L_DIM * C_DIM * 2);
  bf16* Kb = (bf16*)alloc((size_t)L_DIM * C_DIM * 2);
  bf16* Vb = (bf16*)alloc((size_t)L_DIM * C_DIM * 2);
  bf16* Cb = (bf16*)alloc((size_t)L2_DIM * C_DIM * 2);
  bf16* Gb = (bf16*)alloc((size_t)L_DIM * FF_DIM * 2);
  float* S1 = (float*)alloc((size_t)L_DIM * C_DIM * 4);
  float* Ab = (float*)alloc((size_t)L_DIM * C_DIM * 4);
  float* Xb = (float*)alloc((size_t)L_DIM * C_DIM * 4);

  const int nLC = L_DIM * C_DIM;

  auto cvt = [&](const float* src, bf16* dst, int n) {
    cvt_f32_bf16<<<min(cdiv(n, 256), 32768), 256, 0, stream>>>(src, dst, n);
  };
  auto cvtT = [&](const float* src, bf16* dst, int R, int N) {   // [RxN] -> [NxR]
    cvt_transpose_f32_bf16<<<dim3(N / 32, R / 32), 256, 0, stream>>>(src, dst, R, N);
  };
  auto gemm = [&](const bf16* A, const bf16* Bt, const float* bias,
                  float* oF, bf16* oB, int M, int N, int K, int epi) {
    dim3 g(N / GBN, cdiv(M, GBM));
    gemm_bf16_kernel<<<g, 256, 0, stream>>>(A, Bt, bias, oF, oB, M, N, K, epi);
  };

  // transposed bf16 weights (fit in the 192MB L2; tile re-reads are L2 hits)
  cvtT(sa_q_w, w_saq, C_DIM, C_DIM); cvtT(sa_k_w, w_sak, C_DIM, C_DIM);
  cvtT(sa_v_w, w_sav, C_DIM, C_DIM); cvtT(sa_o_w, w_sao, C_DIM, C_DIM);
  cvtT(ca_q_w, w_caq, C_DIM, C_DIM); cvtT(ca_k_w, w_cak, C_DIM, C_DIM);
  cvtT(ca_v_w, w_cav, C_DIM, C_DIM); cvtT(ca_o_w, w_cao, C_DIM, C_DIM);
  cvtT(ffn_w1, w_ff1, C_DIM, FF_DIM);   // [C x FF] -> [FF x C]
  cvtT(ffn_w2, w_ff2, FF_DIM, C_DIM);   // [FF x C] -> [C x FF]
  cvt(ctx, Cb, L2_DIM * C_DIM);

  // ---- self attention ----
  ln_mod_kernel<<<L_DIM, 256, 0, stream>>>(x, e6, mod6, 0, 1, Hb);
  gemm(Hb, w_saq, sa_q_b, S1, nullptr, L_DIM, C_DIM, C_DIM, 0);
  rms_rope_kernel<<<L_DIM, 256, 0, stream>>>(S1, sa_nq, freqs, Qb, 3);
  gemm(Hb, w_sak, sa_k_b, S1, nullptr, L_DIM, C_DIM, C_DIM, 0);
  rms_rope_kernel<<<L_DIM, 256, 0, stream>>>(S1, sa_nk, freqs, Kb, 3);
  gemm(Hb, w_sav, sa_v_b, nullptr, Vb, L_DIM, C_DIM, C_DIM, 1);
  flash_kernel<<<dim3(cdiv(L_DIM, 64), N_HEADS), 128, 0, stream>>>(Qb, Kb, Vb, Ab, L_DIM, L_DIM);
  cvt(Ab, Hb, nLC);
  gemm(Hb, w_sao, sa_o_b, S1, nullptr, L_DIM, C_DIM, C_DIM, 0);
  resid_kernel<<<min(cdiv(nLC, 256), 32768), 256, 0, stream>>>(x, S1, e6, mod6, 2, Xb, nLC);

  // ---- cross attention ----
  cvt(Xb, Hb, nLC);
  gemm(Hb, w_caq, ca_q_b, S1, nullptr, L_DIM, C_DIM, C_DIM, 0);
  rms_rope_kernel<<<L_DIM, 256, 0, stream>>>(S1, ca_nq, freqs, Qb, 1);
  gemm(Cb, w_cak, ca_k_b, S1, nullptr, L2_DIM, C_DIM, C_DIM, 0);
  rms_rope_kernel<<<L2_DIM, 256, 0, stream>>>(S1, ca_nk, freqs, Kb, 1);
  gemm(Cb, w_cav, ca_v_b, nullptr, Vb, L2_DIM, C_DIM, C_DIM, 1);
  flash_kernel<<<dim3(cdiv(L_DIM, 64), N_HEADS), 128, 0, stream>>>(Qb, Kb, Vb, Ab, L_DIM, L2_DIM);
  cvt(Ab, Hb, nLC);
  gemm(Hb, w_cao, ca_o_b, S1, nullptr, L_DIM, C_DIM, C_DIM, 0);
  resid_kernel<<<min(cdiv(nLC, 256), 32768), 256, 0, stream>>>(Xb, S1, e6, mod6, -1, Xb, nLC);

  // ---- FFN ----
  ln_mod_kernel<<<L_DIM, 256, 0, stream>>>(Xb, e6, mod6, 3, 4, Hb);
  gemm(Hb, w_ff1, ffn_b1, nullptr, Gb, L_DIM, FF_DIM, C_DIM, 2);   // fused GELU -> bf16
  gemm(Gb, w_ff2, ffn_b2, S1, nullptr, L_DIM, C_DIM, FF_DIM, 0);
  resid_kernel<<<min(cdiv(nLC, 256), 32768), 256, 0, stream>>>(Xb, S1, e6, mod6, 5, outp, nLC);
}
